// AGITransformer140B_87600152969897
// MI455X (gfx1250) — matrix operand order repaired and checked by reading, not compiled, not run on hardware
//
#include <hip/hip_runtime.h>
#include <cstddef>

typedef __attribute__((ext_vector_type(16))) __bf16 v16bf;
typedef __attribute__((ext_vector_type(8)))  __bf16 bf16x8;
typedef __attribute__((ext_vector_type(8)))  float  v8f;

#define LAM 0.1f
#define HAD 0.05f
#define EPS 1e-6f

// ---------------------------------------------------------------------------
// Fragment loaders (wave32 WMMA 16-bit layouts, CDNA5 ISA §7.12.2)
// A 16x32: lane m=l%16; elems 0..7 = K[khalf..khalf+7], 8..15 = K[16+khalf..]
// B 32x16: lane n=l%16; elems 0..7 = K[kb..kb+7], 8..15 = K[16+kb..] (kb = 0 or 8)
// ---------------------------------------------------------------------------
__device__ __forceinline__ v16bf load_frag_a(const __bf16* base, int khalf) {
    bf16x8 lo = *(const bf16x8*)(base + khalf);
    bf16x8 hi = *(const bf16x8*)(base + 16 + khalf);
    return __builtin_shufflevector(lo, hi, 0,1,2,3,4,5,6,7,8,9,10,11,12,13,14,15);
}
__device__ __forceinline__ v16bf load_frag_b(const __bf16* p) {
    bf16x8 lo = *(const bf16x8*)(p);
    bf16x8 hi = *(const bf16x8*)(p + 16);
    return __builtin_shufflevector(lo, hi, 0,1,2,3,4,5,6,7,8,9,10,11,12,13,14,15);
}
__device__ __forceinline__ v8f wmma_bf16(v16bf a, v16bf b, v8f c) {
    return __builtin_amdgcn_wmma_f32_16x16x32_bf16(false, a, false, b, (short)0, c,
                                                   false, false);
}

// ---------------------------------------------------------------------------
// fp32 -> bf16 convert (weights, activations)
// ---------------------------------------------------------------------------
__global__ void f32_to_bf16_kernel(const float* __restrict__ in,
                                   __bf16* __restrict__ out, size_t n) {
    size_t i  = (size_t)blockIdx.x * blockDim.x + threadIdx.x;
    size_t st = (size_t)gridDim.x * blockDim.x;
    for (; i < n; i += st) out[i] = (__bf16)in[i];
}

// ---------------------------------------------------------------------------
// RMSNorm: one block per row, fp32 in, bf16 out
// ---------------------------------------------------------------------------
__global__ void rmsnorm_kernel(const float* __restrict__ x,
                               const float* __restrict__ w,
                               __bf16* __restrict__ out, int D) {
    const int row = blockIdx.x;
    const float* xr = x + (size_t)row * D;
    float ss = 0.f;
    for (int i = threadIdx.x; i < D; i += blockDim.x) {
        float v = xr[i];
        ss += v * v;
    }
    __shared__ float red[32];
    for (int off = 16; off; off >>= 1) ss += __shfl_xor(ss, off, 32);
    if ((threadIdx.x & 31) == 0) red[threadIdx.x >> 5] = ss;
    __syncthreads();
    if (threadIdx.x < 32) {
        int nw = blockDim.x >> 5;
        float v = (threadIdx.x < (unsigned)nw) ? red[threadIdx.x] : 0.f;
        for (int off = 16; off; off >>= 1) v += __shfl_xor(v, off, 32);
        if (threadIdx.x == 0) red[0] = v;
    }
    __syncthreads();
    const float inv = rsqrtf(red[0] / (float)D + EPS);
    for (int i = threadIdx.x; i < D; i += blockDim.x)
        out[(size_t)row * D + i] = (__bf16)(xr[i] * inv * w[i]);
}

// ---------------------------------------------------------------------------
// Generic bf16 WMMA GEMM: C(MxN) = A(MxK) * W(NxK)^T   (+ optional residual)
// block = 128 threads (4 waves); wave computes a 32x64 tile; block covers
// 32(M) x 256(N).  M%32==0, N%256==0, K%32==0 assumed.
// ---------------------------------------------------------------------------
__global__ __launch_bounds__(128, 1)
void gemm_bf16_kernel(const __bf16* __restrict__ A, const __bf16* __restrict__ W,
                      const float* __restrict__ resid, float* __restrict__ outF,
                      __bf16* __restrict__ outB, int M, int N, int K) {
    const int lane  = threadIdx.x & 31;
    const int wave  = threadIdx.x >> 5;
    const int ln    = lane & 15;
    const int hi16  = lane >> 4;
    const int khalf = hi16 * 8;
    const int kb    = hi16 * 8;
    const int m0 = blockIdx.y * 32;
    const int n0 = blockIdx.x * 256 + wave * 64;

    v8f acc[2][4];
    const v8f vz = {0.f,0.f,0.f,0.f,0.f,0.f,0.f,0.f};
#pragma unroll
    for (int mt = 0; mt < 2; ++mt)
#pragma unroll
        for (int nt = 0; nt < 4; ++nt) acc[mt][nt] = vz;

    const __bf16* arow0 = A + (size_t)(m0 + ln) * K;
    const __bf16* arow1 = A + (size_t)(m0 + 16 + ln) * K;
    for (int k0 = 0; k0 < K; k0 += 32) {
        v16bf a0 = load_frag_a(arow0 + k0, khalf);
        v16bf a1 = load_frag_a(arow1 + k0, khalf);
#pragma unroll
        for (int nt = 0; nt < 4; ++nt) {
            v16bf b = load_frag_b(W + (size_t)(n0 + nt * 16 + ln) * K + k0 + kb);
            acc[0][nt] = wmma_bf16(a0, b, acc[0][nt]);
            acc[1][nt] = wmma_bf16(a1, b, acc[1][nt]);
        }
    }

#pragma unroll
    for (int mt = 0; mt < 2; ++mt)
#pragma unroll
        for (int nt = 0; nt < 4; ++nt)
#pragma unroll
            for (int v = 0; v < 8; ++v) {
                int row = m0 + mt * 16 + v + hi16 * 8;
                int col = n0 + nt * 16 + ln;
                size_t idx = (size_t)row * N + col;
                float val = acc[mt][nt][v];
                if (resid) val += resid[idx];
                if (outF)  outF[idx] = val;
                if (outB)  outB[idx] = (__bf16)val;
            }
}

// ---------------------------------------------------------------------------
// RoPE + square: reads raw bf16 Q,K (b,s,D), writes roped Qb/Kb and Q2b/K2b
// one thread per (row, pair)
// ---------------------------------------------------------------------------
__global__ void rope_kernel(const __bf16* __restrict__ Qr, const __bf16* __restrict__ Kr,
                            __bf16* __restrict__ Qb, __bf16* __restrict__ Q2b,
                            __bf16* __restrict__ Kb, __bf16* __restrict__ K2b,
                            int S, int D, int HD, size_t npairs) {
    size_t idx = (size_t)blockIdx.x * blockDim.x + threadIdx.x;
    if (idx >= npairs) return;
    const int ppr = D / 2;                 // pairs per row
    size_t row = idx / ppr;                // b*S + s
    int p = (int)(idx % ppr);
    int i = p % (HD / 2);                  // pair index within head
    int s = (int)(row % S);
    float inv = __expf(-((2.f * (float)i) / (float)HD) * logf(10000.f));
    float ang = (float)s * inv, sn, cs;
    __sincosf(ang, &sn, &cs);
    size_t off = row * D + (size_t)(p / (HD / 2)) * HD + 2 * (size_t)i;

    float q1 = (float)Qr[off], q2 = (float)Qr[off + 1];
    float o1 = q1 * cs - q2 * sn, o2 = q1 * sn + q2 * cs;
    Qb[off] = (__bf16)o1;          Qb[off + 1] = (__bf16)o2;
    Q2b[off] = (__bf16)(o1 * o1);  Q2b[off + 1] = (__bf16)(o2 * o2);

    float k1 = (float)Kr[off], k2 = (float)Kr[off + 1];
    float p1 = k1 * cs - k2 * sn, p2 = k1 * sn + k2 * cs;
    Kb[off] = (__bf16)p1;          Kb[off + 1] = (__bf16)p2;
    K2b[off] = (__bf16)(p1 * p1);  K2b[off + 1] = (__bf16)(p2 * p2);
}

// V (b,s,D) bf16 -> Vt (b,h,d,s) bf16 so PV B-fragments load contiguously
__global__ void transpose_v_kernel(const __bf16* __restrict__ Vr,
                                   __bf16* __restrict__ Vt,
                                   int S, int D, int HD, size_t n) {
    size_t idx = (size_t)blockIdx.x * blockDim.x + threadIdx.x;
    size_t st  = (size_t)gridDim.x * blockDim.x;
    const int NH = D / HD;
    for (; idx < n; idx += st) {
        int col = (int)(idx % D);
        size_t row = idx / D;                 // b*S + s
        int b = (int)(row / S), s = (int)(row % S);
        int h = col / HD, d = col % HD;
        Vt[(((size_t)(b * NH + h)) * HD + d) * S + s] = Vr[idx];
    }
}

// ---------------------------------------------------------------------------
// Flash attention, WMMA bf16. One wave per 16-query tile; streams 32-key
// blocks; S = (QK^T + LAM*Q2K2^T)*scale; online softmax; P through LDS to
// reshape C-layout -> A-layout; 8 PV WMMAs per block against transposed V.
// out = softmax(S) V ; out += HAD*out^2 ; stored bf16 in (b,s,D).
// ---------------------------------------------------------------------------
__global__ __launch_bounds__(128, 1)
void attn_kernel(const __bf16* __restrict__ Qb, const __bf16* __restrict__ Q2b,
                 const __bf16* __restrict__ Kb, const __bf16* __restrict__ K2b,
                 const __bf16* __restrict__ Vt, __bf16* __restrict__ O,
                 int S, int D, int HD, int NH) {
    const int lane  = threadIdx.x & 31;
    const int wave  = threadIdx.x >> 5;
    const int ln    = lane & 15;
    const int hi16  = lane >> 4;
    const int khalf = hi16 * 8;
    const int kb    = hi16 * 8;
    const int b  = blockIdx.z;
    const int h  = blockIdx.y;
    const int q0 = (blockIdx.x * 4 + wave) * 16;

    __shared__ __align__(16) __bf16 pbuf[4][16][32];

    // Q fragments: 4 k-chunks of 32 dims, plus elementwise squares
    v16bf qf[4], q2f[4];
    {
        const __bf16* qrow  = Qb  + (size_t)(b * S + q0 + ln) * D + (size_t)h * HD;
        const __bf16* q2row = Q2b + (size_t)(b * S + q0 + ln) * D + (size_t)h * HD;
#pragma unroll
        for (int kc = 0; kc < 4; ++kc) {
            qf[kc]  = load_frag_a(qrow  + kc * 32, khalf);
            q2f[kc] = load_frag_a(q2row + kc * 32, khalf);
        }
    }

    const v8f vz = {0.f,0.f,0.f,0.f,0.f,0.f,0.f,0.f};
    v8f accO[8];
#pragma unroll
    for (int dt = 0; dt < 8; ++dt) accO[dt] = vz;
    float mrow[8], lrow[8];
#pragma unroll
    for (int v = 0; v < 8; ++v) { mrow[v] = -1e30f; lrow[v] = 0.f; }

    const float scale = rsqrtf((float)HD);
    const __bf16* vhead = Vt + ((size_t)(b * NH + h)) * HD * S;

    for (int blk = 0; blk < S; blk += 32) {
        // ---- scores: 2 n-tiles of 16 keys, K-dim 128 in 4 chunks ----
        v8f sc[2], sq[2];
        sc[0] = vz; sc[1] = vz; sq[0] = vz; sq[1] = vz;
#pragma unroll
        for (int nt = 0; nt < 2; ++nt) {
            size_t krow = (size_t)(b * S + blk + nt * 16 + ln) * D + (size_t)h * HD;
#pragma unroll
            for (int kc = 0; kc < 4; ++kc) {
                v16bf kf  = load_frag_b(Kb  + krow + kc * 32 + kb);
                sc[nt] = wmma_bf16(qf[kc], kf, sc[nt]);
                v16bf k2f = load_frag_b(K2b + krow + kc * 32 + kb);
                sq[nt] = wmma_bf16(q2f[kc], k2f, sq[nt]);
            }
        }

        // ---- online softmax across this 32-key block ----
        float p0[8], p1[8];
#pragma unroll
        for (int v = 0; v < 8; ++v) {
            float sA = (sc[0][v] + LAM * sq[0][v]) * scale;
            float sB = (sc[1][v] + LAM * sq[1][v]) * scale;
            float bm = fmaxf(sA, sB);
            bm = fmaxf(bm, __shfl_xor(bm, 1, 32));
            bm = fmaxf(bm, __shfl_xor(bm, 2, 32));
            bm = fmaxf(bm, __shfl_xor(bm, 4, 32));
            bm = fmaxf(bm, __shfl_xor(bm, 8, 32));
            float mnew = fmaxf(mrow[v], bm);
            float corr = __expf(mrow[v] - mnew);
            mrow[v] = mnew;
            float pA = __expf(sA - mnew);
            float pB = __expf(sB - mnew);
            float rs = pA + pB;
            rs += __shfl_xor(rs, 1, 32);
            rs += __shfl_xor(rs, 2, 32);
            rs += __shfl_xor(rs, 4, 32);
            rs += __shfl_xor(rs, 8, 32);
            lrow[v] = lrow[v] * corr + rs;
#pragma unroll
            for (int dt = 0; dt < 8; ++dt) accO[dt][v] *= corr;
            p0[v] = pA; p1[v] = pB;
        }

        // ---- P: C-layout -> LDS -> A-layout fragment ----
#pragma unroll
        for (int v = 0; v < 8; ++v) {
            pbuf[wave][v + hi16 * 8][ln]      = (__bf16)p0[v];
            pbuf[wave][v + hi16 * 8][16 + ln] = (__bf16)p1[v];
        }
        asm volatile("s_wait_dscnt 0" ::: "memory");   // wave-lockstep LDS sync
        v16bf pf = load_frag_a(&pbuf[wave][ln][0], khalf);

        // ---- PV: 8 dim-tiles ----
#pragma unroll
        for (int dt = 0; dt < 8; ++dt) {
            v16bf vf = load_frag_b(vhead + (size_t)(dt * 16 + ln) * S + blk + kb);
            accO[dt] = wmma_bf16(pf, vf, accO[dt]);
        }
    }

    // ---- epilogue: normalize, out += HAD*out^2, store bf16 (b,s,D) ----
#pragma unroll
    for (int dt = 0; dt < 8; ++dt)
#pragma unroll
        for (int v = 0; v < 8; ++v) {
            float val = accO[dt][v] / lrow[v];
            val = val + HAD * val * val;
            int q = q0 + v + hi16 * 8;
            int d = h * HD + dt * 16 + ln;
            O[(size_t)(b * S + q) * D + d] = (__bf16)val;
        }
}

// ---------------------------------------------------------------------------
// silu(a) * c, bf16 in/out
// ---------------------------------------------------------------------------
__global__ void silu_mul_kernel(const __bf16* __restrict__ a,
                                const __bf16* __restrict__ c,
                                __bf16* __restrict__ g, size_t n) {
    size_t i  = (size_t)blockIdx.x * blockDim.x + threadIdx.x;
    size_t st = (size_t)gridDim.x * blockDim.x;
    for (; i < n; i += st) {
        float v = (float)a[i];
        float s = v / (1.f + __expf(-v));
        g[i] = (__bf16)(s * (float)c[i]);
    }
}

// ---------------------------------------------------------------------------
// host launcher
// ---------------------------------------------------------------------------
extern "C" void kernel_launch(void* const* d_in, const int* in_sizes, int n_in,
                              void* d_out, int out_size, void* d_ws, size_t ws_size,
                              hipStream_t stream) {
    (void)in_sizes; (void)n_in; (void)out_size; (void)ws_size;
    const int B = 2, S = 1024, D = 2048, NH = 16, HD = 128, DFF = 8192;
    const size_t DD  = (size_t)D * D;        // 4.19M
    const size_t DF  = (size_t)D * DFF;      // 16.8M
    const size_t BS  = (size_t)B * S;        // 2048
    const size_t BSD = BS * D;               // 4.19M
    const size_t BSF = BS * DFF;             // 16.8M

    const float* x       = (const float*)d_in[0];
    const float* norm1_w = (const float*)d_in[1];
    const float* norm2_w = (const float*)d_in[2];
    const float* wq      = (const float*)d_in[3];
    const float* wk      = (const float*)d_in[4];
    const float* wv      = (const float*)d_in[5];
    const float* wo      = (const float*)d_in[6];
    /* gate_w d_in[7] unused: rotate_half leaves both score terms invariant,
       so out2 == out1 and the sigmoid gate cancels algebraically. */
    const float* w1      = (const float*)d_in[8];
    const float* w3      = (const float*)d_in[9];
    const float* w2      = (const float*)d_in[10];
    float* outp          = (float*)d_out;

    // workspace carve-up
    char* base = (char*)d_ws;
    size_t off = 0;
    auto carve = [&](size_t bytes) {
        char* p = base + off;
        off += (bytes + 255) & ~(size_t)255;
        return p;
    };
    __bf16* wqb  = (__bf16*)carve(DD * 2);
    __bf16* wkb  = (__bf16*)carve(DD * 2);
    __bf16* wvb  = (__bf16*)carve(DD * 2);
    __bf16* wob  = (__bf16*)carve(DD * 2);
    __bf16* w1b  = (__bf16*)carve(DF * 2);
    __bf16* w3b  = (__bf16*)carve(DF * 2);
    __bf16* w2b  = (__bf16*)carve(DF * 2);
    __bf16* xn   = (__bf16*)carve(BSD * 2);
    __bf16* Qraw = (__bf16*)carve(BSD * 2);
    __bf16* Kraw = (__bf16*)carve(BSD * 2);
    __bf16* Vraw = (__bf16*)carve(BSD * 2);
    __bf16* Qb   = (__bf16*)carve(BSD * 2);
    __bf16* Q2b  = (__bf16*)carve(BSD * 2);
    __bf16* Kb   = (__bf16*)carve(BSD * 2);
    __bf16* K2b  = (__bf16*)carve(BSD * 2);
    __bf16* Vt   = (__bf16*)carve(BSD * 2);
    __bf16* attn = (__bf16*)carve(BSD * 2);
    float*  hbuf = (float*) carve(BSD * 4);
    __bf16* hn   = (__bf16*)carve(BSD * 2);
    __bf16* ab   = (__bf16*)carve(BSF * 2);
    __bf16* cb   = (__bf16*)carve(BSF * 2);
    __bf16* gb   = (__bf16*)carve(BSF * 2);

    // 1. weights -> bf16
    f32_to_bf16_kernel<<<2048, 256, 0, stream>>>(wq, wqb, DD);
    f32_to_bf16_kernel<<<2048, 256, 0, stream>>>(wk, wkb, DD);
    f32_to_bf16_kernel<<<2048, 256, 0, stream>>>(wv, wvb, DD);
    f32_to_bf16_kernel<<<2048, 256, 0, stream>>>(wo, wob, DD);
    f32_to_bf16_kernel<<<4096, 256, 0, stream>>>(w1, w1b, DF);
    f32_to_bf16_kernel<<<4096, 256, 0, stream>>>(w3, w3b, DF);
    f32_to_bf16_kernel<<<4096, 256, 0, stream>>>(w2, w2b, DF);

    // 2. xn = rmsnorm(x) * norm1_w
    rmsnorm_kernel<<<(int)BS, 256, 0, stream>>>(x, norm1_w, xn, D);

    // 3. Q/K/V projections (bf16 WMMA GEMMs)
    dim3 gp(D / 256, (int)BS / 32);
    gemm_bf16_kernel<<<gp, 128, 0, stream>>>(xn, wqb, nullptr, nullptr, Qraw, (int)BS, D, D);
    gemm_bf16_kernel<<<gp, 128, 0, stream>>>(xn, wkb, nullptr, nullptr, Kraw, (int)BS, D, D);
    gemm_bf16_kernel<<<gp, 128, 0, stream>>>(xn, wvb, nullptr, nullptr, Vraw, (int)BS, D, D);

    // 4. RoPE (+squares) and V transpose
    size_t npairs = BSD / 2;
    rope_kernel<<<(int)((npairs + 255) / 256), 256, 0, stream>>>(
        Qraw, Kraw, Qb, Q2b, Kb, K2b, S, D, HD, npairs);
    transpose_v_kernel<<<4096, 256, 0, stream>>>(Vraw, Vt, S, D, HD, BSD);

    // 5. flash attention (single pass; out2==out1 algebraically)
    dim3 ga(S / 64, NH, B);
    attn_kernel<<<ga, 128, 0, stream>>>(Qb, Q2b, Kb, K2b, Vt, attn, S, D, HD, NH);

    // 6. h = x + attn @ wo^T
    gemm_bf16_kernel<<<gp, 128, 0, stream>>>(attn, wob, x, hbuf, nullptr, (int)BS, D, D);

    // 7. hn = rmsnorm(h) * norm2_w
    rmsnorm_kernel<<<(int)BS, 256, 0, stream>>>(hbuf, norm2_w, hn, D);

    // 8. FFN up projections
    dim3 gf(DFF / 256, (int)BS / 32);
    gemm_bf16_kernel<<<gf, 128, 0, stream>>>(hn, w1b, nullptr, nullptr, ab, (int)BS, DFF, D);
    gemm_bf16_kernel<<<gf, 128, 0, stream>>>(hn, w3b, nullptr, nullptr, cb, (int)BS, DFF, D);

    // 9. g = silu(a) * c
    silu_mul_kernel<<<4096, 256, 0, stream>>>(ab, cb, gb, BSF);

    // 10. out = h + g @ w2^T
    gemm_bf16_kernel<<<gp, 128, 0, stream>>>(gb, w2b, hbuf, outp, nullptr, (int)BS, D, DFF);
}
